// MultiHeadPNCAAttention_88802743812461
// MI455X (gfx1250) — matrix operand order repaired and checked by reading, not compiled, not run on hardware
//
#include <hip/hip_runtime.h>

// ---------------------------------------------------------------------------
// Problem constants (from reference)
// ---------------------------------------------------------------------------
#define Bc  2
#define LXc 2048
#define LHc 1024
#define DMc 1024
#define Hc  16
#define Dc  64

typedef __attribute__((ext_vector_type(16))) __bf16 v16bf;
typedef __attribute__((ext_vector_type(8)))  float  v8f;

union BFfrag { unsigned int u[8]; v16bf v; };
union U4     { uint4 q; unsigned int u[4]; unsigned short s[8]; };

__device__ __forceinline__ unsigned short f32_to_bf16_rte(float f) {
  unsigned int u = __float_as_uint(f);
  u += 0x7FFFu + ((u >> 16) & 1u);          // round to nearest even
  return (unsigned short)(u >> 16);
}

__device__ __forceinline__ unsigned int pack_bf16x2(float lo, float hi) {
  return (unsigned int)f32_to_bf16_rte(lo) |
         ((unsigned int)f32_to_bf16_rte(hi) << 16);
}

// ---------------------------------------------------------------------------
// Generic bf16 WMMA GEMM:  C = alpha * A(MxK) * B(KxN) [+bias] [+adds]
//   TM/TN    : workgroup tile (8 waves of 32 lanes, each wave 32x32)
//   BT       : B stored transposed, element (k,n) = Bm[n*ldb + k]
//   OUT_BF16 : write bf16 (raw u16) instead of f32
//   A_F32    : A operand is f32 in memory, converted to bf16 on LDS fill
//   USE_MASK : boolean mask [M,N]; nonzero -> write -inf (pre-softmax scores)
//   HAS_BIAS : add bias[n]
//   NADD     : number of residual adds (0/1/2), addressed [m*ldc + n]
// Requirements: M % TM == 0, N % TN == 0, K % 32 == 0 (grid covers exactly).
// Batch offset for tensor T at z = blockIdx.z:  (z/inner)*sTo + (z%inner)*sTi
// Register + LDS double buffering: fetch tile t+1 into VGPRs while WMMAs
// consume tile t from LDS; deposit registers into the ping-pong buffer after
// compute so the loadcnt wait overlaps the matrix work.
// ---------------------------------------------------------------------------
template<int TM, int TN, bool BT, bool OUT_BF16, bool A_F32, bool USE_MASK,
         bool HAS_BIAS, int NADD>
__global__ __launch_bounds__(256)
void wmma_gemm_kernel(const void* __restrict__ Av,
                      const unsigned short* __restrict__ Bm,
                      const float* __restrict__ bias,
                      const unsigned char* __restrict__ mask,
                      const float* __restrict__ add0,
                      const float* __restrict__ add1,
                      void* __restrict__ Cv,
                      int M, int N, int K, int lda, int ldb, int ldc,
                      int inner,
                      long long sAo, long long sAi,
                      long long sBo, long long sBi,
                      long long sCo, long long sCi,
                      long long sMo, long long sMi,
                      float alpha)
{
  constexpr int WM  = TM / 32;             // waves along M
  constexpr int WN  = TN / 32;             // waves along N
  static_assert(WM * WN == 8, "need 8 waves");
  constexpr int LD  = 32 + 8;              // LDS row stride (80B, 16B-aligned)
  constexpr int ACH = TM / 64;             // A b128 chunks per thread
  constexpr int BCH = TN / 64;             // B b128 chunks per thread

  __shared__ __align__(16) unsigned short As[2][TM][LD];   // [buf][m][k]
  __shared__ __align__(16) unsigned short Bs[2][TN][LD];   // [buf][n][k]

  const int z  = blockIdx.z;
  const int zo = z / inner, zi = z % inner;

  const unsigned short* A16 = (const unsigned short*)Av;
  const float*          A32 = (const float*)Av;
  const long long aoff = zo * sAo + zi * sAi;
  const unsigned short* Bp = Bm + zo * sBo + zi * sBi;
  const long long coff = zo * sCo + zi * sCi;
  const unsigned char* Mp = USE_MASK ? (mask + zo * sMo + zi * sMi) : nullptr;

  const int m0   = blockIdx.y * TM;
  const int n0   = blockIdx.x * TN;
  const int tid  = threadIdx.x;
  const int lane = tid & 31;
  const int wave = tid >> 5;
  const int wm   = wave % WM;
  const int wn   = wave / WM;
  const int lg   = lane >> 4;              // lane group (0/1)
  const int ll   = lane & 15;

  // Per-thread tile-fill coordinates (constant across k-steps)
  int ar[ACH];
  const int ac = (tid & 3) * 8;
#pragma unroll
  for (int it = 0; it < ACH; ++it) ar[it] = (tid >> 2) + 64 * it;

  int bn[BCH], bk[BCH];                    // BT: row n, k-chunk | !BT: k, n-chunk
#pragma unroll
  for (int j = 0; j < BCH; ++j) {
    const int cid = tid + 256 * j;
    if (BT) { bn[j] = cid >> 2;        bk[j] = (cid & 3) * 8; }
    else    { bn[j] = cid / (TN / 8);  bk[j] = (cid % (TN / 8)) * 8; }
  }

  // Register staging for the in-flight tile
  uint4  ra[ACH];
  float4 raf[ACH][2];
  uint4  rb[BCH];

  const int nk = K / 32;

  auto fetch_tile = [&](int kt) {
    const int k0 = kt * 32;
#pragma unroll
    for (int it = 0; it < ACH; ++it) {
      const long long gbase = aoff + (long long)(m0 + ar[it]) * lda + (k0 + ac);
      if (A_F32) {
        const float* src = &A32[gbase];
        raf[it][0] = *(const float4*)(src);
        raf[it][1] = *(const float4*)(src + 4);
        if (kt + 1 < nk) __builtin_prefetch(src + 32, 0, 3);
      } else {
        ra[it] = *(const uint4*)&A16[gbase];
        if (kt + 1 < nk) __builtin_prefetch(&A16[gbase + 32], 0, 3);
      }
    }
#pragma unroll
    for (int j = 0; j < BCH; ++j) {
      if (BT)
        rb[j] = *(const uint4*)&Bp[(long long)(n0 + bn[j]) * ldb + (k0 + bk[j])];
      else
        rb[j] = *(const uint4*)&Bp[(long long)(k0 + bn[j]) * ldb + (n0 + bk[j])];
    }
  };

  auto store_tile = [&](int buf) {
#pragma unroll
    for (int it = 0; it < ACH; ++it) {
      if (A_F32) {
        U4 w;
        w.u[0] = pack_bf16x2(raf[it][0].x, raf[it][0].y);
        w.u[1] = pack_bf16x2(raf[it][0].z, raf[it][0].w);
        w.u[2] = pack_bf16x2(raf[it][1].x, raf[it][1].y);
        w.u[3] = pack_bf16x2(raf[it][1].z, raf[it][1].w);
        *(uint4*)&As[buf][ar[it]][ac] = w.q;
      } else {
        *(uint4*)&As[buf][ar[it]][ac] = ra[it];
      }
    }
#pragma unroll
    for (int j = 0; j < BCH; ++j) {
      if (BT) {
        *(uint4*)&Bs[buf][bn[j]][bk[j]] = rb[j];
      } else {
        U4 w; w.q = rb[j];
#pragma unroll
        for (int e = 0; e < 8; ++e) Bs[buf][bk[j] + e][bn[j]] = w.s[e];
      }
    }
  };

  v8f acc[2][2] = {};

  // Prologue: tile 0 into buffer 0
  fetch_tile(0);
  store_tile(0);

  for (int kt = 0; kt < nk; ++kt) {
    __syncthreads();
    const int cur = kt & 1;

    if (kt + 1 < nk) fetch_tile(kt + 1);   // loads in flight during compute

    // ---- fragments per documented wave32 VGPR layouts ----
    BFfrag af[2], bf[2];
#pragma unroll
    for (int mi = 0; mi < 2; ++mi) {
      const int row = wm * 32 + mi * 16 + ll;
#pragma unroll
      for (int i = 0; i < 8; ++i) {
        const int kb = (i < 4) ? (2 * i + 8 * lg) : (16 + 2 * (i - 4) + 8 * lg);
        af[mi].u[i] = *(const unsigned int*)&As[cur][row][kb];
      }
    }
#pragma unroll
    for (int ni = 0; ni < 2; ++ni) {
      const int col = wn * 32 + ni * 16 + ll;
#pragma unroll
      for (int i = 0; i < 8; ++i) {
        const int kb = (i < 4) ? (2 * i + 8 * lg) : (16 + 2 * (i - 4) + 8 * lg);
        bf[ni].u[i] = *(const unsigned int*)&Bs[cur][col][kb];
      }
    }
#pragma unroll
    for (int mi = 0; mi < 2; ++mi)
#pragma unroll
      for (int ni = 0; ni < 2; ++ni)
        acc[mi][ni] = __builtin_amdgcn_wmma_f32_16x16x32_bf16(
            false, af[mi].v, false, bf[ni].v, (short)0, acc[mi][ni], false, false);

    if (kt + 1 < nk) store_tile(cur ^ 1);  // wait lands after the WMMAs
  }

  // ---- epilogue: C layout lane ll = col, m_local = vgpr + 8*lg ----
  unsigned short* C16 = (unsigned short*)Cv;
  float*          C32 = (float*)Cv;
#pragma unroll
  for (int mi = 0; mi < 2; ++mi) {
#pragma unroll
    for (int ni = 0; ni < 2; ++ni) {
#pragma unroll
      for (int v = 0; v < 8; ++v) {
        const int m = m0 + wm * 32 + mi * 16 + 8 * lg + v;
        const int n = n0 + wn * 32 + ni * 16 + ll;
        float val = acc[mi][ni][v] * alpha;
        if (HAS_BIAS)  val += bias[n];
        if (NADD >= 1) val += add0[(long long)m * ldc + n];
        if (NADD >= 2) val += add1[(long long)m * ldc + n];
        if (USE_MASK && Mp[(long long)m * N + n]) val = -__builtin_inff();
        const long long ci = coff + (long long)m * ldc + n;
        if (OUT_BF16) C16[ci] = f32_to_bf16_rte(val);
        else          C32[ci] = val;
      }
    }
  }
}

// ---------------------------------------------------------------------------
// LayerNorm over last dim (cols), output bf16 raw u16. One block per row.
// ---------------------------------------------------------------------------
__global__ __launch_bounds__(256)
void layernorm_bf16_kernel(const float* __restrict__ x,
                           const float* __restrict__ gain,
                           const float* __restrict__ beta,
                           unsigned short* __restrict__ out, int cols)
{
  const long long row = blockIdx.x;
  const float* p = x + row * cols;
  __shared__ float red[256];
  const int tid = threadIdx.x;

  float s = 0.f;
  for (int c = tid; c < cols; c += 256) s += p[c];
  red[tid] = s; __syncthreads();
  for (int st = 128; st > 0; st >>= 1) {
    if (tid < st) red[tid] += red[tid + st];
    __syncthreads();
  }
  const float mean = red[0] / cols;
  __syncthreads();

  float v = 0.f;
  for (int c = tid; c < cols; c += 256) { float d = p[c] - mean; v += d * d; }
  red[tid] = v; __syncthreads();
  for (int st = 128; st > 0; st >>= 1) {
    if (tid < st) red[tid] += red[tid + st];
    __syncthreads();
  }
  const float rstd = rsqrtf(red[0] / cols + 1e-6f);

  for (int c = tid; c < cols; c += 256)
    out[row * cols + c] = f32_to_bf16_rte((p[c] - mean) * rstd * gain[c] + beta[c]);
}

// ---------------------------------------------------------------------------
// Elementwise f32 -> bf16 raw (vectorized x4; n must be multiple of 4)
// ---------------------------------------------------------------------------
__global__ __launch_bounds__(256)
void f32_to_bf16_kernel(const float* __restrict__ in,
                        unsigned short* __restrict__ out, long long n4)
{
  const long long i = (long long)blockIdx.x * 256 + threadIdx.x;
  if (i < n4) {
    float4 f = *(const float4*)&in[i * 4];
    unsigned int w0 = pack_bf16x2(f.x, f.y);
    unsigned int w1 = pack_bf16x2(f.z, f.w);
    *(uint2*)&out[i * 4] = make_uint2(w0, w1);
  }
}

// ---------------------------------------------------------------------------
// [B,L,S,H,D] (GEMM output, row-major) -> S separate [B,H,L,D] buffers (bf16)
// Processes 8 contiguous d at a time (b128 load + b128 store).
// ---------------------------------------------------------------------------
__global__ __launch_bounds__(256)
void reshape_heads_kernel(const unsigned short* __restrict__ in,
                          unsigned short* __restrict__ d0,
                          unsigned short* __restrict__ d1,
                          unsigned short* __restrict__ d2,
                          int L, int S, long long total8)
{
  const long long i = (long long)blockIdx.x * 256 + threadIdx.x;
  if (i >= total8) return;
  long long t = i;
  const int d8 = (int)(t % (Dc / 8)); t /= (Dc / 8);
  const int h  = (int)(t % Hc);       t /= Hc;
  const int s  = (int)(t % S);        t /= S;
  const int l  = (int)(t % L);
  const int b  = (int)(t / L);
  const long long o = ((((long long)(b * Hc + h) * L) + l) * Dc) + d8 * 8;
  unsigned short* dst = (s == 0) ? d0 : (s == 1) ? d1 : d2;
  *(uint4*)&dst[o] = *(const uint4*)&in[i * 8];
}

// ---------------------------------------------------------------------------
// In-place row softmax over `cols`. One block per row.
// ---------------------------------------------------------------------------
__global__ __launch_bounds__(256)
void softmax_rows_kernel(float* __restrict__ attn, int cols)
{
  float* p = attn + (long long)blockIdx.x * cols;
  __shared__ float red[256];
  const int tid = threadIdx.x;

  float mx = -__builtin_inff();
  for (int c = tid; c < cols; c += 256) mx = fmaxf(mx, p[c]);
  red[tid] = mx; __syncthreads();
  for (int st = 128; st > 0; st >>= 1) {
    if (tid < st) red[tid] = fmaxf(red[tid], red[tid + st]);
    __syncthreads();
  }
  mx = red[0]; __syncthreads();

  float s = 0.f;
  for (int c = tid; c < cols; c += 256) s += __expf(p[c] - mx);
  red[tid] = s; __syncthreads();
  for (int st = 128; st > 0; st >>= 1) {
    if (tid < st) red[tid] += red[tid + st];
    __syncthreads();
  }
  const float inv = 1.0f / red[0];

  for (int c = tid; c < cols; c += 256) p[c] = __expf(p[c] - mx) * inv;
}

// ---------------------------------------------------------------------------
// Host launcher
// ---------------------------------------------------------------------------
extern "C" void kernel_launch(void* const* d_in, const int* in_sizes, int n_in,
                              void* d_out, int out_size, void* d_ws, size_t ws_size,
                              hipStream_t stream)
{
  (void)in_sizes; (void)n_in; (void)out_size; (void)ws_size;

  const float*         x      = (const float*)d_in[0];
  const float*         hmem   = (const float*)d_in[1];
  const unsigned char* mask_x = (const unsigned char*)d_in[2];
  const unsigned char* mask_h = (const unsigned char*)d_in[3];
  const float*         ln_g   = (const float*)d_in[4];
  const float*         ln_b   = (const float*)d_in[5];
  const float*         w_qkv  = (const float*)d_in[6];
  const float*         b_qkv  = (const float*)d_in[7];
  const float*         w_fcx  = (const float*)d_in[8];
  const float*         b_fcx  = (const float*)d_in[9];
  const float*         w_hkv  = (const float*)d_in[10];
  const float*         b_hkv  = (const float*)d_in[11];
  const float*         w_fch  = (const float*)d_in[12];
  const float*         b_fch  = (const float*)d_in[13];

  // Output tuple layout: output [B,LX,DM] | attn_x [B,H,LX,LX] | attn_h [B,H,LX,LH]
  float* outp   = (float*)d_out;
  float* attn_x = outp + (long long)Bc * LXc * DMc;
  float* attn_h = attn_x + (long long)Bc * Hc * LXc * LXc;

  // Workspace layout (u16 elements)
  unsigned short* xn    = (unsigned short*)d_ws;                      // [4096,1024]
  unsigned short* hb    = xn    + (long long)Bc * LXc * DMc;          // [2048,1024]
  unsigned short* wqkvb = hb    + (long long)Bc * LHc * DMc;          // [1024,3072]
  unsigned short* whkvb = wqkvb + (long long)DMc * 3 * Hc * Dc;       // [1024,2048]
  unsigned short* wfcxb = whkvb + (long long)DMc * 2 * Hc * Dc;       // [1024,1024]
  unsigned short* wfchb = wfcxb + (long long)DMc * DMc;               // [1024,1024]
  unsigned short* qbuf  = wfchb + (long long)DMc * DMc;               // [B,H,LX,D]
  unsigned short* kxbuf = qbuf  + (long long)Bc * Hc * LXc * Dc;
  unsigned short* vxbuf = kxbuf + (long long)Bc * Hc * LXc * Dc;
  unsigned short* hkbuf = vxbuf + (long long)Bc * Hc * LXc * Dc;      // [B,H,LH,D]
  unsigned short* hvbuf = hkbuf + (long long)Bc * Hc * LHc * Dc;
  unsigned short* qkvst = hvbuf + (long long)Bc * Hc * LHc * Dc;      // [4096,3072]
  unsigned short* hkvst = qkvst + (long long)Bc * LXc * 3 * Hc * Dc;  // [2048,2048]
  // After reshape consumes the staging buffers, alias them:
  unsigned short* outx  = qkvst;                                      // [4096,1024] bf16
  unsigned short* outh  = outx + (long long)Bc * LXc * DMc;           // [4096,1024] bf16
  float*          fx    = (float*)(outh + (long long)Bc * LXc * DMc); // [4096,1024] f32

  const dim3 blk(256);

  // 1) LayerNorm(x) -> bf16
  layernorm_bf16_kernel<<<Bc * LXc, blk, 0, stream>>>(x, ln_g, ln_b, xn, DMc);

  // 2) bf16 converts for h + weights
  {
    long long n4;
    n4 = (long long)Bc * LHc * DMc / 4;
    f32_to_bf16_kernel<<<(unsigned)((n4 + 255) / 256), blk, 0, stream>>>(hmem, hb, n4);
    n4 = (long long)DMc * 3 * Hc * Dc / 4;
    f32_to_bf16_kernel<<<(unsigned)((n4 + 255) / 256), blk, 0, stream>>>(w_qkv, wqkvb, n4);
    n4 = (long long)DMc * 2 * Hc * Dc / 4;
    f32_to_bf16_kernel<<<(unsigned)((n4 + 255) / 256), blk, 0, stream>>>(w_hkv, whkvb, n4);
    n4 = (long long)DMc * DMc / 4;
    f32_to_bf16_kernel<<<(unsigned)((n4 + 255) / 256), blk, 0, stream>>>(w_fcx, wfcxb, n4);
    f32_to_bf16_kernel<<<(unsigned)((n4 + 255) / 256), blk, 0, stream>>>(w_fch, wfchb, n4);
  }

  // 3) QKV GEMM: [4096,1024] x [1024,3072] + b_qkv -> bf16 staging
  wmma_gemm_kernel<64, 128, false, true, false, false, true, 0>
      <<<dim3(3072 / 128, 4096 / 64, 1), blk, 0, stream>>>(
      xn, wqkvb, b_qkv, nullptr, nullptr, nullptr, qkvst,
      4096, 3072, 1024, 1024, 3072, 3072, 1,
      0, 0, 0, 0, 0, 0, 0, 0, 1.0f);

  // 4) HKV GEMM: [2048,1024] x [1024,2048] + b_hkv -> bf16 staging
  wmma_gemm_kernel<64, 128, false, true, false, false, true, 0>
      <<<dim3(2048 / 128, 2048 / 64, 1), blk, 0, stream>>>(
      hb, whkvb, b_hkv, nullptr, nullptr, nullptr, hkvst,
      2048, 2048, 1024, 1024, 2048, 2048, 1,
      0, 0, 0, 0, 0, 0, 0, 0, 1.0f);

  // 5) Reshape to [B,H,L,D]
  {
    long long tq8 = (long long)Bc * LXc * 3 * Hc * Dc / 8;
    reshape_heads_kernel<<<(unsigned)((tq8 + 255) / 256), blk, 0, stream>>>(
        qkvst, qbuf, kxbuf, vxbuf, LXc, 3, tq8);
    long long th8 = (long long)Bc * LHc * 2 * Hc * Dc / 8;
    reshape_heads_kernel<<<(unsigned)((th8 + 255) / 256), blk, 0, stream>>>(
        hkvst, hkbuf, hvbuf, nullptr, LHc, 2, th8);
  }

  // 6) Scores: S = Q * K^T * (1/8), masked, f32 -> d_out attn regions
  wmma_gemm_kernel<64, 128, true, false, false, true, false, 0>
      <<<dim3(LXc / 128, LXc / 64, Bc * Hc), blk, 0, stream>>>(
      qbuf, kxbuf, nullptr, mask_x, nullptr, nullptr, attn_x,
      LXc, LXc, Dc, Dc, Dc, LXc, Hc,
      (long long)Hc * LXc * Dc, (long long)LXc * Dc,
      (long long)Hc * LXc * Dc, (long long)LXc * Dc,
      (long long)Hc * LXc * LXc, (long long)LXc * LXc,
      (long long)LXc * LXc, 0, 0.125f);

  wmma_gemm_kernel<64, 128, true, false, false, true, false, 0>
      <<<dim3(LHc / 128, LXc / 64, Bc * Hc), blk, 0, stream>>>(
      qbuf, hkbuf, nullptr, mask_h, nullptr, nullptr, attn_h,
      LXc, LHc, Dc, Dc, Dc, LHc, Hc,
      (long long)Hc * LXc * Dc, (long long)LXc * Dc,
      (long long)Hc * LHc * Dc, (long long)LHc * Dc,
      (long long)Hc * LXc * LHc, (long long)LXc * LHc,
      (long long)LXc * LHc, 0, 0.125f);

  // 7) Row softmax in place (outputs are the attention matrices)
  softmax_rows_kernel<<<Bc * Hc * LXc, blk, 0, stream>>>(attn_x, LXc);
  softmax_rows_kernel<<<Bc * Hc * LXc, blk, 0, stream>>>(attn_h, LHc);

  // 8) attn @ V -> [B,LX,H*D] bf16 (A is f32, converted on load; head-
  //    interleaved C).  Tile 128x64 since N = D = 64.
  wmma_gemm_kernel<128, 64, false, true, true, false, false, 0>
      <<<dim3(Dc / 64, LXc / 128, Bc * Hc), blk, 0, stream>>>(
      attn_x, vxbuf, nullptr, nullptr, nullptr, nullptr, outx,
      LXc, Dc, LXc, LXc, Dc, Hc * Dc, Hc,
      (long long)Hc * LXc * LXc, (long long)LXc * LXc,
      (long long)Hc * LXc * Dc, (long long)LXc * Dc,
      (long long)LXc * DMc, (long long)Dc,
      0, 0, 1.0f);

  wmma_gemm_kernel<128, 64, false, true, true, false, false, 0>
      <<<dim3(Dc / 64, LXc / 128, Bc * Hc), blk, 0, stream>>>(
      attn_h, hvbuf, nullptr, nullptr, nullptr, nullptr, outh,
      LXc, Dc, LHc, LHc, Dc, Hc * Dc, Hc,
      (long long)Hc * LXc * LHc, (long long)LXc * LHc,
      (long long)Hc * LHc * Dc, (long long)LHc * Dc,
      (long long)LXc * DMc, (long long)Dc,
      0, 0, 1.0f);

  // 9) Output projections: fx = out_x @ w_fcx + b_fcx (f32 scratch)
  wmma_gemm_kernel<64, 128, false, false, false, false, true, 0>
      <<<dim3(1024 / 128, 4096 / 64, 1), blk, 0, stream>>>(
      outx, wfcxb, b_fcx, nullptr, nullptr, nullptr, fx,
      4096, 1024, 1024, 1024, 1024, 1024, 1,
      0, 0, 0, 0, 0, 0, 0, 0, 1.0f);

  // 10) Final: d_out = out_h @ w_fch + b_fch + fx + x (residual), f32
  wmma_gemm_kernel<64, 128, false, false, false, false, true, 2>
      <<<dim3(1024 / 128, 4096 / 64, 1), blk, 0, stream>>>(
      outh, wfchb, b_fch, nullptr, fx, x, outp,
      4096, 1024, 1024, 1024, 1024, 1024, 1,
      0, 0, 0, 0, 0, 0, 0, 0, 1.0f);
}
